// TPLinear_6725918786080
// MI455X (gfx1250) — 8-bit matrix operand order repaired and checked by reading, compiled, not run on hardware
//
#include <hip/hip_runtime.h>
#include <hip/hip_bf16.h>
#include <stdint.h>

typedef __attribute__((ext_vector_type(16))) int   v16i;
typedef __attribute__((ext_vector_type(8)))  float v8f;

#define FP8_MAX 448.0f


__device__ __forceinline__ float bf16_round(float v) {
  uint32_t u = __float_as_uint(v);
  u = (u + 0x7FFFu + ((u >> 16) & 1u)) & 0xFFFF0000u;
  return __uint_as_float(u);
}

__device__ __forceinline__ float clamp448(float v) {
  return fminf(fmaxf(v, -FP8_MAX), FP8_MAX);
}

#if defined(__has_builtin)
#if __has_builtin(__builtin_amdgcn_cvt_pk_fp8_f32)
#define HAVE_CVT_PK_FP8 1
#endif
#endif

__device__ __forceinline__ uint32_t e4m3_sw(float f) {
  uint32_t u = __float_as_uint(f);
  uint32_t s = (u >> 24) & 0x80u;
  float a = fabsf(f);
  if (!(a >= 0.0009765625f)) return s;
  if (a >= FP8_MAX) return s | 0x7Eu;
  int e; float m = frexpf(a, &e);
  int E = e - 1;
  uint32_t code;
  if (E < -6) {
    code = (uint32_t)rintf(a * 512.0f);
    if (code > 8u) code = 8u;
  } else {
    uint32_t q = (uint32_t)rintf(m * 16.0f);
    if (q == 16u) { q = 8u; ++E; }
    if (E > 8) return s | 0x7Eu;
    code = ((uint32_t)(E + 7) << 3) | (q - 8u);
  }
  return s | code;
}

__device__ __forceinline__ uint32_t pack4_e4m3(float a, float b, float c, float d) {
#ifdef HAVE_CVT_PK_FP8
  int lo = __builtin_amdgcn_cvt_pk_fp8_f32(a, b, 0, false);
  int r  = __builtin_amdgcn_cvt_pk_fp8_f32(c, d, lo, true);
  return (uint32_t)r;
#else
  return e4m3_sw(a) | (e4m3_sw(b) << 8) | (e4m3_sw(c) << 16) | (e4m3_sw(d) << 24);
#endif
}

__device__ __forceinline__ void async_load_b128(uint32_t lds_off, const void* gptr) {
  asm volatile("global_load_async_to_lds_b128 %0, %1, off"
               :: "v"(lds_off), "v"((uint64_t)(uintptr_t)gptr)
               : "memory");
}


__global__ void k_init(unsigned int* amax_bits) {
  if (threadIdx.x == 0) *amax_bits = 0u;
}


__global__ void k_amax(const float4* __restrict__ x4,
                       unsigned int* __restrict__ amax_bits, int n4) {
  float m = 0.0f;
  for (int i = blockIdx.x * blockDim.x + threadIdx.x; i < n4;
       i += gridDim.x * blockDim.x) {
    float4 v = x4[i];
    m = fmaxf(m, fabsf(bf16_round(v.x)));
    m = fmaxf(m, fabsf(bf16_round(v.y)));
    m = fmaxf(m, fabsf(bf16_round(v.z)));
    m = fmaxf(m, fabsf(bf16_round(v.w)));
  }
#pragma unroll
  for (int off = 16; off > 0; off >>= 1)
    m = fmaxf(m, __shfl_xor(m, off, 32));
  __shared__ float red[8];
  const int lane = threadIdx.x & 31, wv = threadIdx.x >> 5;
  if (lane == 0) red[wv] = m;
  __syncthreads();
  if (threadIdx.x == 0) {
    float mm = red[0];
#pragma unroll
    for (int i = 1; i < 8; ++i) mm = fmaxf(mm, red[i]);
    atomicMax(amax_bits, __float_as_uint(mm));
  }
}


__global__ void k_quant_x(const float4* __restrict__ x4,
                          uint4* __restrict__ xq16,
                          const unsigned int* __restrict__ amax_bits, int n16) {
  const int i = blockIdx.x * blockDim.x + threadIdx.x;
  if (i >= n16) return;
  const float amaxv = fmaxf(__uint_as_float(*amax_bits), 1e-12f);
  const float inv = FP8_MAX / amaxv;
  uint4 r;
  uint32_t* rp = (uint32_t*)&r;
#pragma unroll
  for (int j = 0; j < 4; ++j) {
    float4 v = x4[i * 4 + j];
    rp[j] = pack4_e4m3(clamp448(bf16_round(v.x) * inv),
                       clamp448(bf16_round(v.y) * inv),
                       clamp448(bf16_round(v.z) * inv),
                       clamp448(bf16_round(v.w) * inv));
  }
  xq16[i] = r;
}


__global__ void k_quant_w(const float4* __restrict__ w4,
                          uint4* __restrict__ wq16, int n16) {
  const int i = blockIdx.x * blockDim.x + threadIdx.x;
  if (i >= n16) return;
  uint4 r;
  uint32_t* rp = (uint32_t*)&r;
#pragma unroll
  for (int j = 0; j < 4; ++j) {
    float4 v = w4[i * 4 + j];
    rp[j] = pack4_e4m3(v.x, v.y, v.z, v.w);
  }
  wq16[i] = r;
}


__global__ void __launch_bounds__(256)
k_gemm_fp8(const unsigned char* __restrict__ xq,
           const unsigned char* __restrict__ wq,
           const unsigned int* __restrict__ amax_bits,
           const float* __restrict__ wscale,
           const float* __restrict__ bias,
           __hip_bfloat16* __restrict__ out,
           int M, int N, int K) {
  __shared__ uint4 smemB[2][8 * 128];

  const int lane = threadIdx.x & 31;
  const int wave = threadIdx.x >> 5;
  const int half_ = lane >> 4;
  const int mrow = lane & 15;

  const int n0 = blockIdx.x * 128;
  const int m0 = blockIdx.y * 128;
  const int m_wave = m0 + wave * 16;

  const int stage_n = threadIdx.x >> 1;
  const int stage_kh = (threadIdx.x & 1) << 6;

  v8f acc[8];
#pragma unroll
  for (int n = 0; n < 8; ++n) acc[n] = (v8f){0, 0, 0, 0, 0, 0, 0, 0};

  const unsigned char* arow_base = xq + (size_t)(m_wave + mrow) * K + half_ * 16;
  const unsigned char* wrow_base = wq + (size_t)(n0 + stage_n) * K + stage_kh;

  uint32_t lds_off[2][4];
#pragma unroll
  for (int b = 0; b < 2; ++b)
#pragma unroll
    for (int i = 0; i < 4; ++i) {
      const int c = stage_kh + i * 16;
      const int g = c >> 5, h = (c >> 4) & 1;
      lds_off[b][i] =
          (uint32_t)(uintptr_t)&smemB[b][(g * 2 + h) * 128 + stage_n];
    }

  const int nk = K >> 7;

#pragma unroll
  for (int i = 0; i < 4; ++i)
    async_load_b128(lds_off[0][i], wrow_base + i * 16);

  for (int it = 0; it < nk; ++it) {
    const int ks = it << 7;
    const int cur = it & 1;

    if (it + 1 < nk) {
      const unsigned char* wsrc = wrow_base + (ks + 128);
#pragma unroll
      for (int i = 0; i < 4; ++i)
        async_load_b128(lds_off[cur ^ 1][i], wsrc + i * 16);
      asm volatile("s_wait_asynccnt 0x4" ::: "memory");
    } else {
      asm volatile("s_wait_asynccnt 0x0" ::: "memory");
    }
    __syncthreads();

    union { v16i v; int2 d[8]; } afrag;
    const unsigned char* arow = arow_base + ks;
#pragma unroll
    for (int j = 0; j < 8; ++j)
      afrag.d[j] = *(const int2*)(arow + (j >> 1) * 32 + (j & 1) * 8);

#pragma unroll
    for (int nsub = 0; nsub < 8; ++nsub) {
      union { v16i v; uint4 q[4]; } bfrag;
      const int ncol = nsub * 16 + mrow;
#pragma unroll
      for (int g = 0; g < 4; ++g)
        bfrag.q[g] = smemB[cur][(g * 2 + half_) * 128 + ncol];

      acc[nsub] = __builtin_amdgcn_wmma_f32_16x16x128_fp8_fp8(
          afrag.v, bfrag.v, (short)0, acc[nsub], false, false);
    }
    __syncthreads();
  }

  const float amaxv = fmaxf(__uint_as_float(*amax_bits), 1e-12f);
  const float act_scale = amaxv / FP8_MAX;
#pragma unroll
  for (int nsub = 0; nsub < 8; ++nsub) {
    const int col = n0 + nsub * 16 + mrow;
    const float sc = act_scale * wscale[col];
    const float bs = bias[col];
#pragma unroll
    for (int i = 0; i < 8; ++i) {
      const int row = m_wave + half_ * 8 + i;
      out[(size_t)row * N + col] = __float2bfloat16(acc[nsub][i] * sc + bs);
    }
  }
}


extern "C" void kernel_launch(void* const* d_in, const int* in_sizes, int n_in,
                              void* d_out, int out_size, void* d_ws, size_t ws_size,
                              hipStream_t stream) {
  const float* x      = (const float*)d_in[0];
  const float* w      = (const float*)d_in[1];
  const float* wscale = (const float*)d_in[2];
  const float* bias   = (const float*)d_in[3];
  __hip_bfloat16* out = (__hip_bfloat16*)d_out;

  const int N = in_sizes[2];
  const int K = in_sizes[1] / N;
  const int M = in_sizes[0] / K;

  unsigned int*  amax_bits = (unsigned int*)d_ws;
  unsigned char* xq = (unsigned char*)d_ws + 256;
  unsigned char* wq = xq + (size_t)M * K;

  k_init<<<1, 1, 0, stream>>>(amax_bits);

  const int n4x = (M * K) / 4;
  k_amax<<<1024, 256, 0, stream>>>((const float4*)x, amax_bits, n4x);

  const int n16x = (M * K) / 16;
  k_quant_x<<<(n16x + 255) / 256, 256, 0, stream>>>(
      (const float4*)x, (uint4*)xq, amax_bits, n16x);

  const int n16w = (N * K) / 16;
  k_quant_w<<<(n16w + 255) / 256, 256, 0, stream>>>(
      (const float4*)w, (uint4*)wq, n16w);

  dim3 grid(N / 128, M / 128);
  k_gemm_fp8<<<grid, 256, 0, stream>>>(xq, wq, amax_bits, wscale, bias, out,
                                       M, N, K);
}
